// SR_GNN_83708912599529
// MI455X (gfx1250) — matrix operand order repaired and checked by reading, not compiled, not run on hardware
//
#include <hip/hip_runtime.h>
#include <hip/hip_bf16.h>
#include <math.h>

typedef __bf16 bf16_t;
typedef __attribute__((ext_vector_type(16))) __bf16 v16bf;
typedef __attribute__((ext_vector_type(8)))  float  v8f;

// ---------------------------------------------------------------------------
// WMMA fragment helpers (wave32; A/B use contiguous k = 16*half + s)
// A fragment: 16x32 bf16 tile, LDS row-major [m][k]. lane m = lane%16,
//   half = lane/16; K window = half*16 .. +15 packed 2-per-VGPR.
// B fragment: 32x16 bf16 tile, LDS stored transposed [n][k]. lane n = lane%16,
//   K window = (lane/16)*16 .. +15 packed 2-per-VGPR.
// C/D: v8f, element r -> (m = r + 8*(lane/16), n = lane%16).
// ---------------------------------------------------------------------------
static __device__ __forceinline__ v16bf frag_a_lds(const bf16_t* base, int lda, int lane) {
  const int m = lane & 15;
  const int half = lane >> 4;
  const bf16_t* row = base + m * lda + half * 16;
  v16bf a;
#pragma unroll
  for (int j = 0; j < 4; ++j) {
    a[2 * j]     = row[2 * j];
    a[2 * j + 1] = row[2 * j + 1];
    a[8 + 2 * j]     = row[8 + 2 * j];
    a[8 + 2 * j + 1] = row[8 + 2 * j + 1];
  }
  return a;
}

static __device__ __forceinline__ v16bf frag_b_lds(const bf16_t* base, int ldb, int lane) {
  const int n = lane & 15;
  const int kh = (lane >> 4) * 16;
  const bf16_t* row = base + n * ldb + kh;
  v16bf b;
#pragma unroll
  for (int j = 0; j < 8; ++j) {
    b[2 * j]     = row[2 * j];
    b[2 * j + 1] = row[2 * j + 1];
  }
  return b;
}

static __device__ __forceinline__ v8f wmma_bf16(v16bf a, v16bf b, v8f c) {
  return __builtin_amdgcn_wmma_f32_16x16x32_bf16(false, a, false, b, (short)0, c,
                                                 false, false);
}

// ---------------------------------------------------------------------------
// Generic bf16-WMMA GEMM: C[M,N] = A[M,K] * B + bias
//   TB == 0 : B is row-major [K][N]
//   TB == 1 : B is row-major [N][K] (i.e. we multiply by B^T)
// Block: 256 threads (8 waves). Tile 64(M) x 128(N), K-step 32.
// Each wave owns four 16x16 output tiles (one A fragment reused 4x).
// Double-buffered LDS with register staging: one barrier per K-step,
// global loads for step s+1 overlap the WMMAs of step s.
// ---------------------------------------------------------------------------
template <int TB>
__global__ __launch_bounds__(256) void gemm_bf16(const float* __restrict__ A,
                                                 const float* __restrict__ Bm,
                                                 const float* __restrict__ bias,
                                                 float* __restrict__ C, int M,
                                                 int N, int K) {
  constexpr int LDA = 40;
  constexpr int LDB = 40;
  __shared__ bf16_t As[2][64 * LDA];
  __shared__ bf16_t Bs[2][128 * LDB];  // [n][k]

  const int t = threadIdx.x;
  const int lane = t & 31;
  const int wave = t >> 5;
  const int m0 = blockIdx.y * 64;
  const int n0 = blockIdx.x * 128;
  const int mt = wave & 3;
  const int ntb = (wave >> 2) * 4;

  v8f acc[4] = {};
  float4 ra[2];
  float4 rb[4];

  auto load_global = [&](int kk) {
#pragma unroll
    for (int it = 0; it < 2; ++it) {
      int idx = t + it * 256;
      int r = idx >> 3, c4 = (idx & 7) * 4;
      int gm = m0 + r;
      ra[it] = make_float4(0.f, 0.f, 0.f, 0.f);
      if (gm < M) {
        const float* p = A + (size_t)gm * K + kk + c4;
        ra[it] = *(const float4*)p;
        if (kk + 32 < K) __builtin_prefetch(p + 32, 0, 0);
      }
    }
    if (TB) {
#pragma unroll
      for (int it = 0; it < 4; ++it) {
        int idx = t + it * 256;
        int r = idx >> 3, c4 = (idx & 7) * 4;
        int gn = n0 + r;
        rb[it] = make_float4(0.f, 0.f, 0.f, 0.f);
        if (gn < N) {
          const float* p = Bm + (size_t)gn * K + kk + c4;
          rb[it] = *(const float4*)p;
          if (kk + 32 < K) __builtin_prefetch(p + 32, 0, 0);
        }
      }
    } else {
#pragma unroll
      for (int it = 0; it < 4; ++it) {
        int idx = t + it * 256;
        int kr = idx >> 5, c4 = (idx & 31) * 4;
        int gn = n0 + c4;
        const float* src = Bm + (size_t)(kk + kr) * N + gn;
        float4 v = make_float4(0.f, 0.f, 0.f, 0.f);
        if (gn + 3 < N) {
          v = *(const float4*)src;
        } else {
          if (gn + 0 < N) v.x = src[0];
          if (gn + 1 < N) v.y = src[1];
          if (gn + 2 < N) v.z = src[2];
          if (gn + 3 < N) v.w = src[3];
        }
        rb[it] = v;
      }
    }
  };

  auto store_lds = [&](int buf) {
#pragma unroll
    for (int it = 0; it < 2; ++it) {
      int idx = t + it * 256;
      int r = idx >> 3, c4 = (idx & 7) * 4;
      bf16_t* d = As[buf] + r * LDA + c4;
      d[0] = (bf16_t)ra[it].x; d[1] = (bf16_t)ra[it].y;
      d[2] = (bf16_t)ra[it].z; d[3] = (bf16_t)ra[it].w;
    }
    if (TB) {
#pragma unroll
      for (int it = 0; it < 4; ++it) {
        int idx = t + it * 256;
        int r = idx >> 3, c4 = (idx & 7) * 4;
        bf16_t* d = Bs[buf] + r * LDB + c4;
        d[0] = (bf16_t)rb[it].x; d[1] = (bf16_t)rb[it].y;
        d[2] = (bf16_t)rb[it].z; d[3] = (bf16_t)rb[it].w;
      }
    } else {
#pragma unroll
      for (int it = 0; it < 4; ++it) {
        int idx = t + it * 256;
        int kr = idx >> 5, c4 = (idx & 31) * 4;
        Bs[buf][(c4 + 0) * LDB + kr] = (bf16_t)rb[it].x;
        Bs[buf][(c4 + 1) * LDB + kr] = (bf16_t)rb[it].y;
        Bs[buf][(c4 + 2) * LDB + kr] = (bf16_t)rb[it].z;
        Bs[buf][(c4 + 3) * LDB + kr] = (bf16_t)rb[it].w;
      }
    }
  };

  load_global(0);
  store_lds(0);
  const int nsteps = K >> 5;
  for (int s = 0; s < nsteps; ++s) {
    const int buf = s & 1;
    const bool has_next = (s + 1) < nsteps;
    __syncthreads();
    if (has_next) load_global((s + 1) * 32);
    v16bf af = frag_a_lds(As[buf] + mt * 16 * LDA, LDA, lane);
#pragma unroll
    for (int j = 0; j < 4; ++j) {
      v16bf bf = frag_b_lds(Bs[buf] + (ntb + j) * 16 * LDB, LDB, lane);
      acc[j] = wmma_bf16(af, bf, acc[j]);
    }
    if (has_next) store_lds(buf ^ 1);
  }

  const int half = lane >> 4;
#pragma unroll
  for (int j = 0; j < 4; ++j) {
    int n = n0 + (ntb + j) * 16 + (lane & 15);
    if (n >= N) continue;
    float bv = bias ? bias[n] : 0.f;
#pragma unroll
    for (int r = 0; r < 8; ++r) {
      int m = m0 + mt * 16 + r + 8 * half;
      if (m < M) C[(size_t)m * N + n] = acc[j][r] + bv;
    }
  }
}

// ---------------------------------------------------------------------------
// Batched adjacency message passing: per batch b,
//   out[:, 0:256]   = A[:, 0:32]  @ hin   (32x32 @ 32x256)
//   out[:, 256:512] = A[:, 32:64] @ hout
// K = 32 => exactly one WMMA per 16x16 output tile. One workgroup per batch.
// ---------------------------------------------------------------------------
__global__ __launch_bounds__(256) void adj_msg(const float* __restrict__ Aadj,
                                               const float* __restrict__ hin,
                                               const float* __restrict__ hout,
                                               float* __restrict__ out) {
  constexpr int LDA = 68;
  constexpr int LDB = 36;
  __shared__ bf16_t Asm[32 * LDA];
  __shared__ bf16_t Bin[256 * LDB];
  __shared__ bf16_t Bout[256 * LDB];

  const int b = blockIdx.x;
  const int t = threadIdx.x;
  const int lane = t & 31, wave = t >> 5;
  const float* Ab = Aadj + (size_t)b * 32 * 64;
  const float* hinb = hin + (size_t)b * 32 * 256;
  const float* houtb = hout + (size_t)b * 32 * 256;
  float* outb = out + (size_t)b * 32 * 512;

  // A: 32x64 fp32 -> bf16 LDS (row-major [m][col])
#pragma unroll
  for (int it = 0; it < 2; ++it) {
    int idx = t + it * 256;
    int r = idx >> 4, c4 = (idx & 15) * 4;
    float4 v = *(const float4*)(Ab + r * 64 + c4);
    bf16_t* d = Asm + r * LDA + c4;
    d[0] = (bf16_t)v.x; d[1] = (bf16_t)v.y;
    d[2] = (bf16_t)v.z; d[3] = (bf16_t)v.w;
  }
  // hin/hout: 32x256 each -> transposed [n][k] bf16 LDS
#pragma unroll
  for (int it = 0; it < 8; ++it) {
    int idx = t + it * 256;
    int k = idx >> 6, n4 = (idx & 63) * 4;
    float4 vi = *(const float4*)(hinb + k * 256 + n4);
    float4 vo = *(const float4*)(houtb + k * 256 + n4);
    Bin[(n4 + 0) * LDB + k] = (bf16_t)vi.x;
    Bin[(n4 + 1) * LDB + k] = (bf16_t)vi.y;
    Bin[(n4 + 2) * LDB + k] = (bf16_t)vi.z;
    Bin[(n4 + 3) * LDB + k] = (bf16_t)vi.w;
    Bout[(n4 + 0) * LDB + k] = (bf16_t)vo.x;
    Bout[(n4 + 1) * LDB + k] = (bf16_t)vo.y;
    Bout[(n4 + 2) * LDB + k] = (bf16_t)vo.z;
    Bout[(n4 + 3) * LDB + k] = (bf16_t)vo.w;
  }
  __syncthreads();

  // 64 jobs = (half in/out, 2 m-tiles, 16 n-tiles); 8 per wave.
  for (int q = 0; q < 8; ++q) {
    int jb = wave * 8 + q;
    int half = jb >> 5;
    int mt = (jb >> 4) & 1;
    int nt = jb & 15;
    v16bf af = frag_a_lds(Asm + mt * 16 * LDA + half * 32, LDA, lane);
    const bf16_t* Bsel = half ? Bout : Bin;
    v16bf bf = frag_b_lds(Bsel + nt * 16 * LDB, LDB, lane);
    v8f acc = {};
    acc = wmma_bf16(af, bf, acc);
    int n = half * 256 + nt * 16 + (lane & 15);
    int hl = lane >> 4;
#pragma unroll
    for (int r = 0; r < 8; ++r) {
      int m = mt * 16 + r + 8 * hl;
      outb[m * 512 + n] = acc[r];
    }
  }
}

// ---------------------------------------------------------------------------
// Elementwise / gather kernels
// ---------------------------------------------------------------------------
__global__ void gather_emb(const float* __restrict__ emb,
                           const int* __restrict__ items,
                           float* __restrict__ h, int total) {
  int i = blockIdx.x * blockDim.x + threadIdx.x;
  if (i >= total) return;
  int row = i >> 8, c = i & 255;
  h[i] = emb[(size_t)items[row] * 256 + c];
}

__global__ void gru_fuse(const float* __restrict__ gi,
                         const float* __restrict__ gh, float* __restrict__ h,
                         int total) {
  int i = blockIdx.x * blockDim.x + threadIdx.x;
  if (i >= total) return;
  int r = i >> 8, c = i & 255;
  const float* gir = gi + (size_t)r * 768;
  const float* ghr = gh + (size_t)r * 768;
  float ir = gir[c], iz = gir[256 + c], in_ = gir[512 + c];
  float hr = ghr[c], hz = ghr[256 + c], hn = ghr[512 + c];
  float rg = 1.f / (1.f + expf(-(ir + hr)));
  float zg = 1.f / (1.f + expf(-(iz + hz)));
  float ng = tanhf(in_ + rg * hn);
  float ho = h[i];
  h[i] = ng + zg * (ho - ng);
}

__global__ void gather_seq(const float* __restrict__ h,
                           const int* __restrict__ alias,
                           float* __restrict__ hs, int total) {
  int i = blockIdx.x * blockDim.x + threadIdx.x;
  if (i >= total) return;
  int bl = i >> 8, c = i & 255;
  int b = bl / 50;
  hs[i] = h[((size_t)b * 32 + alias[bl]) * 256 + c];
}

__global__ __launch_bounds__(256) void tail_gather(const float* __restrict__ hs,
                                                   const int* __restrict__ mask,
                                                   float* __restrict__ htail) {
  __shared__ int tail;
  int b = blockIdx.x, t = threadIdx.x;
  if (t == 0) {
    int s = 0;
    for (int l = 0; l < 50; ++l) s += mask[b * 50 + l];
    tail = s - 1;
  }
  __syncthreads();
  htail[b * 256 + t] = hs[((size_t)b * 50 + tail) * 256 + t];
}

__global__ __launch_bounds__(256) void attn_pool(const float* __restrict__ q1,
                                                 const float* __restrict__ q2,
                                                 const float* __restrict__ hs,
                                                 const int* __restrict__ mask,
                                                 const float* __restrict__ fc3,
                                                 float* __restrict__ se) {
  __shared__ float red[256];
  int b = blockIdx.x, t = threadIdx.x;
  float q1v = q1[b * 256 + t];
  float w3 = fc3[t];
  float acc = 0.f;
  for (int l = 0; l < 50; ++l) {
    int bl = b * 50 + l;
    float q2v = q2[(size_t)bl * 256 + t];
    float sg = 1.f / (1.f + expf(-(q1v + q2v)));
    red[t] = sg * w3;
    __syncthreads();
#pragma unroll
    for (int s = 128; s > 0; s >>= 1) {
      if (t < s) red[t] += red[t + s];
      __syncthreads();
    }
    float alpha = red[0];
    acc += alpha * (float)mask[bl] * hs[(size_t)bl * 256 + t];
    __syncthreads();
  }
  se[b * 256 + t] = acc;
}

// ---------------------------------------------------------------------------
// Launcher
// ---------------------------------------------------------------------------
extern "C" void kernel_launch(void* const* d_in, const int* in_sizes, int n_in,
                              void* d_out, int out_size, void* d_ws,
                              size_t ws_size, hipStream_t stream) {
  const float* Aadj = (const float*)d_in[0];
  const int* items = (const int*)d_in[1];
  const int* alias = (const int*)d_in[2];
  const int* mask = (const int*)d_in[3];
  const float* emb = (const float*)d_in[4];
  const float* W_in = (const float*)d_in[5];
  const float* b_in = (const float*)d_in[6];
  const float* W_out = (const float*)d_in[7];
  const float* b_out = (const float*)d_in[8];
  const float* W_ih = (const float*)d_in[9];
  const float* b_ih = (const float*)d_in[10];
  const float* W_hh = (const float*)d_in[11];
  const float* b_hh = (const float*)d_in[12];
  const float* fc1_w = (const float*)d_in[13];
  const float* fc1_b = (const float*)d_in[14];
  const float* fc2_w = (const float*)d_in[15];
  const float* fc2_b = (const float*)d_in[16];
  const float* fc3_w = (const float*)d_in[17];
  float* out = (float*)d_out;

  const int M = 512 * 32;  // 16384 node rows
  char* ws = (char*)d_ws;
  size_t off = 0;
  auto alloc = [&](size_t bytes) -> float* {
    float* p = (float*)(ws + off);
    off = (off + bytes + 255) & ~(size_t)255;
    return p;
  };

  float* h = alloc((size_t)M * 256 * 4);      // node states (persist)
  float* hin = alloc((size_t)M * 256 * 4);    // h@W_in+b
  float* hout = alloc((size_t)M * 256 * 4);   // h@W_out+b
  float* acat = alloc((size_t)M * 512 * 4);   // [m_in | m_out]
  float* gi = alloc((size_t)M * 768 * 4);
  float* gh = alloc((size_t)M * 768 * 4);
  // Phase-2 overlay (GRU scratch is dead after gru_fuse):
  float* hs = hin;                  // [B,L,H] 26.2MB over hin+hout (32MB)
  float* q2 = acat;                 // 26.2MB over acat (32MB)
  float* htail = gi;                // 512KB
  float* q1 = gi + 512 * 256;       // 512KB
  float* semb = gi + 2 * 512 * 256; // 512KB

  const int thr = 256;

  gather_emb<<<(M * 256 + thr - 1) / thr, thr, 0, stream>>>(emb, items, h,
                                                            M * 256);

  dim3 gHH((256 + 127) / 128, M / 64);
  gemm_bf16<0><<<gHH, thr, 0, stream>>>(h, W_in, b_in, hin, M, 256, 256);
  gemm_bf16<0><<<gHH, thr, 0, stream>>>(h, W_out, b_out, hout, M, 256, 256);

  adj_msg<<<512, thr, 0, stream>>>(Aadj, hin, hout, acat);

  dim3 gGate((768 + 127) / 128, M / 64);
  gemm_bf16<0><<<gGate, thr, 0, stream>>>(acat, W_ih, b_ih, gi, M, 768, 512);
  gemm_bf16<0><<<gGate, thr, 0, stream>>>(h, W_hh, b_hh, gh, M, 768, 256);

  gru_fuse<<<(M * 256 + thr - 1) / thr, thr, 0, stream>>>(gi, gh, h, M * 256);

  gather_seq<<<(512 * 50 * 256 + thr - 1) / thr, thr, 0, stream>>>(
      h, alias, hs, 512 * 50 * 256);
  tail_gather<<<512, thr, 0, stream>>>(hs, mask, htail);

  gemm_bf16<0><<<dim3(2, 8), thr, 0, stream>>>(htail, fc1_w, fc1_b, q1, 512,
                                               256, 256);
  gemm_bf16<0><<<dim3(2, 400), thr, 0, stream>>>(hs, fc2_w, fc2_b, q2, 25600,
                                                 256, 256);

  attn_pool<<<512, thr, 0, stream>>>(q1, q2, hs, mask, fc3_w, semb);

  // scores = seq_emb @ emb[1:].T  (M=512, N=100000, K=256; B transposed)
  gemm_bf16<1><<<dim3((100000 + 127) / 128, 512 / 64), thr, 0, stream>>>(
      semb, emb + 256, nullptr, out, 512, 100000, 256);
}
